// Multi_head_attention_71554155151423
// MI455X (gfx1250) — matrix operand order repaired and checked by reading, not compiled, not run on hardware
//
#include <hip/hip_runtime.h>
#include <hip/hip_bf16.h>
#include <math.h>
#include <stdint.h>

// ---------------- problem constants ----------------
#define BB   4
#define LD   2048
#define LE   2048
#define HID  768
#define NH   12
#define DK   64
#define NEGV (-1e10f)
#define EPS  1e-8f
// 1/sqrt(768)
#define SCALE 0.036084391824351615f

typedef _Float16 f16;
typedef __attribute__((ext_vector_type(16))) _Float16 v16h;
typedef __attribute__((ext_vector_type(8)))  _Float16 v8h;
typedef __attribute__((ext_vector_type(8)))  float    v8f;

// gfx1250 async global->LDS path via inline asm (signature-free, per ISA
// CDNA5 ISA / CDNA5 ISA VGLOBAL encoding).
#if defined(__gfx1250__)
#define USE_ASYNC_LDS 1
#endif

#if defined(USE_ASYNC_LDS)
__device__ __forceinline__ void wait_async0() {
#if __has_builtin(__builtin_amdgcn_s_wait_asynccnt)
  __builtin_amdgcn_s_wait_asynccnt(0);
#else
  asm volatile("s_wait_asynccnt 0" ::: "memory");
#endif
}
__device__ __forceinline__ void wait_ds0() {
#if __has_builtin(__builtin_amdgcn_s_wait_dscnt)
  __builtin_amdgcn_s_wait_dscnt(0);
#else
  asm volatile("s_wait_dscnt 0" ::: "memory");
#endif
}
// GLOBAL_LOAD_ASYNC_TO_LDS_B128: per-lane 16B global -> LDS copy (ASYNCcnt).
// VDST = 32-bit LDS byte address (flat pointer low 32 bits == LDS offset per
// the aperture mapping), VADDR = 64-bit global address, SADDR = off (NULL).
__device__ __forceinline__ void async_copy_b128(const f16* g, const f16* l) {
  const unsigned           lds = (unsigned)(size_t)l;
  const unsigned long long ga  = (unsigned long long)(size_t)g;
  asm volatile("global_load_async_to_lds_b128 %0, %1, off"
               :: "v"(lds), "v"(ga)
               : "memory");
}
#endif

// ---------------- WMMA helpers ----------------
__device__ __forceinline__ v8f wmma_f16(v16h a, v16h b, v8f c) {
  // D = A(16x32 f16) * B(32x16 f16) + C(16x16 f32)
  return __builtin_amdgcn_wmma_f32_16x16x32_f16(
      /*neg_a=*/false, a, /*neg_b=*/false, b,
      /*c_mod=*/(short)0, c, /*reuse_a=*/false, /*reuse_b=*/false);
}

// A fragment: 16x32 (MxK) f16, row-major source with `stride` halves per row.
// ISA layout: lanes 0-15 -> M=0..15, K groups {0..7,16..23};
//             lanes 16-31 -> same M, K groups {8..15,24..31}.
__device__ __forceinline__ v16h load_a(const f16* base, int stride) {
  const int lane = threadIdx.x & 31;
  const int m    = lane & 15;
  const int kg   = (lane >> 4) * 8;
  const f16* p = base + (size_t)m * stride;
  v8h lo = *(const v8h*)(p + kg);        // K = kg .. kg+7
  v8h hi = *(const v8h*)(p + 16 + kg);   // K = 16+kg .. 16+kg+7
  v16h a;
#pragma unroll
  for (int j = 0; j < 8; ++j) { a[j] = lo[j]; a[j + 8] = hi[j]; }
  return a;
}

// B fragment: 32x16 (KxN) f16 where B[k][n] = SRC[n][k] (SRC row-major with
// `stride` halves per row, i.e. we multiply by SRC^T).
// ISA layout: lanes 0-15 -> N=0..15 holding K {0..7,16..23}; lanes 16-31 -> K {8..15,24..31}.
__device__ __forceinline__ v16h load_b(const f16* base, int stride) {
  const int lane = threadIdx.x & 31;
  const int n    = lane & 15;
  const int ks   = (lane >> 4) * 8;
  const f16* p = base + (size_t)n * stride + ks;
  v8h lo = *(const v8h*)(p);
  v8h hi = *(const v8h*)(p + 16);
  v16h b;
#pragma unroll
  for (int j = 0; j < 8; ++j) { b[j] = lo[j]; b[j + 8] = hi[j]; }
  return b;
}

// ---------------- fp32 -> f16 convert ----------------
__global__ void __launch_bounds__(256) cvt_f32_f16(const float* __restrict__ in,
                                                   f16* __restrict__ out, int n) {
  int i = blockIdx.x * 256 + threadIdx.x;
  if (i < n) out[i] = (f16)in[i];
}

// ---------------- projection GEMM  Y = X * W^T + bias ----------------
// MODE 0: store f16 with strided-head reorder (col j -> (j%12)*64 + j/12)  [Q,K]
// MODE 1: store f16 transposed per head:  Vt[((b*NH+h)*DK + d)*L + le]     [V]
// MODE 2: store f32:  Yf = val + bias + resid                               [out proj]
template <int MODE>
__global__ void __launch_bounds__(256)
proj_gemm(const f16* __restrict__ X, const f16* __restrict__ W,
          const float* __restrict__ bias, const float* __restrict__ resid,
          f16* __restrict__ Y16, float* __restrict__ Yf, int L) {
  const int wave = threadIdx.x >> 5;
  const int lane = threadIdx.x & 31;
  const int row0 = blockIdx.x * 16;                 // 16 rows of flattened [B*L]
  const int n0   = blockIdx.y * 128 + wave * 16;    // 16 output columns

  const f16* Abase = X + (size_t)row0 * HID;
  const f16* Bbase = W + (size_t)n0 * HID;

  v8f c = {};
#pragma unroll 4
  for (int k = 0; k < HID; k += 32) {
    __builtin_prefetch(Abase + k + 128, 0, 1);   // global_prefetch_b8
    __builtin_prefetch(Bbase + k + 128, 0, 1);
    v16h a = load_a(Abase + k, HID);
    v16h b = load_b(Bbase + k, HID);
    c = wmma_f16(a, b, c);
  }

  const int ncol = lane & 15;
  const int mb   = (lane >> 4) * 8;
  const int j    = n0 + ncol;
  const float bj = bias[j];

#pragma unroll
  for (int r = 0; r < 8; ++r) {
    const int row = row0 + mb + r;     // flattened batch*seq row
    float val = c[r] + bj;
    if (MODE == 0) {
      const int col = (j % NH) * DK + (j / NH);
      Y16[(size_t)row * HID + col] = (f16)val;
    } else if (MODE == 1) {
      const int h  = j % NH;
      const int d  = j / NH;
      const int b_ = row / L;
      const int le = row % L;
      Y16[(((size_t)(b_ * NH + h) * DK + d) * L) + le] = (f16)val;
    } else {
      Yf[(size_t)row * HID + j] = val + resid[(size_t)row * HID + j];
    }
  }
}

// ---------------- flash attention: one wave per (b, h, 16-query tile) ----------------
__global__ void __launch_bounds__(256)
attn_kernel(const f16* __restrict__ Qc, const f16* __restrict__ Kc,
            const f16* __restrict__ Vt, const unsigned char* __restrict__ mask,
            f16* __restrict__ A16) {
  __shared__ __align__(16) f16 Pt[8][16][32];   // per-wave P tile (16 q x 32 k)
#if defined(USE_ASYNC_LDS)
  __shared__ __align__(16) f16 Kt[8][32][64];   // per-wave K tile (32 k x 64 d)
#endif

  const int wave = threadIdx.x >> 5;
  const int lane = threadIdx.x & 31;
  const int wid  = blockIdx.x * 8 + wave;
  const int qt   = wid % (LD / 16);
  const int bh   = wid / (LD / 16);
  const int b    = bh / NH;
  const int h    = bh % NH;
  const int q0   = qt * 16;

  const int ncol = lane & 15;
  const int mb   = (lane >> 4) * 8;

  // Q fragments: 16 x 64, split into two K=32 pieces
  const f16* Qbase = Qc + ((size_t)(b * LD + q0)) * HID + h * DK;
  const v16h qa0 = load_a(Qbase, HID);
  const v16h qa1 = load_a(Qbase + 32, HID);

  float mrow[8], lrow[8];
  v8f O[4] = {};
#pragma unroll
  for (int r = 0; r < 8; ++r) { mrow[r] = -INFINITY; lrow[r] = 0.0f; }

  for (int kc = 0; kc < LE; kc += 32) {
    const f16* Kb0 = Kc + ((size_t)(b * LE + kc)) * HID + h * DK;

#if defined(USE_ASYNC_LDS)
    // Stage the 32x64 K tile into per-wave LDS via the async data path.
    // 4KB tile = 8 x (32 lanes x b128). Drain pending DS reads of the tile
    // first, then issue async copies and wait on ASYNCcnt.
    wait_ds0();
#pragma unroll
    for (int s = 0; s < 8; ++s) {
      const int row = s * 4 + (lane >> 3);
      const int seg = (lane & 7) * 8;            // halves
      async_copy_b128(Kb0 + (size_t)row * HID + seg, &Kt[wave][row][seg]);
    }
    wait_async0();
    const f16* KA   = &Kt[wave][0][0];
    const int kstr  = 64;
#else
    const f16* KA   = Kb0;
    const int kstr  = HID;
#endif

    // ---- S = Q * K^T, 16 x 32 in two 16x16 tiles ----
    v8f s0 = {};
    s0 = wmma_f16(qa0, load_b(KA, kstr), s0);
    s0 = wmma_f16(qa1, load_b(KA + 32, kstr), s0);
    v8f s1 = {};
    s1 = wmma_f16(qa0, load_b(KA + 16 * kstr, kstr), s1);
    s1 = wmma_f16(qa1, load_b(KA + 16 * kstr + 32, kstr), s1);

    // ---- online softmax (rows live across 16-lane halves) ----
    float p0[8], p1[8], alpha[8];
#pragma unroll
    for (int r = 0; r < 8; ++r) {
      const unsigned char* mp =
          mask + ((size_t)(b * LD + q0 + mb + r)) * LE + kc;
      float a0 = s0[r] * SCALE;
      float a1 = s1[r] * SCALE;
      if (mp[ncol])      a0 = NEGV;
      if (mp[16 + ncol]) a1 = NEGV;
      float t = fmaxf(a0, a1);
#pragma unroll
      for (int msk = 1; msk < 16; msk <<= 1) t = fmaxf(t, __shfl_xor(t, msk, 32));
      const float nm = fmaxf(mrow[r], t);
      const float e0 = __expf(a0 - nm);
      const float e1 = __expf(a1 - nm);
      float rs = e0 + e1;
#pragma unroll
      for (int msk = 1; msk < 16; msk <<= 1) rs += __shfl_xor(rs, msk, 32);
      const float al = __expf(mrow[r] - nm);
      lrow[r] = lrow[r] * al + rs;
      mrow[r] = nm;
      alpha[r] = al;
      p0[r] = e0;
      p1[r] = e1;
    }
#pragma unroll
    for (int r = 0; r < 8; ++r) {
      O[0][r] *= alpha[r]; O[1][r] *= alpha[r];
      O[2][r] *= alpha[r]; O[3][r] *= alpha[r];
    }

    // ---- transpose P through LDS: C layout -> A-fragment layout ----
#pragma unroll
    for (int r = 0; r < 8; ++r) {
      Pt[wave][mb + r][ncol]      = (f16)p0[r];
      Pt[wave][mb + r][16 + ncol] = (f16)p1[r];
    }
    const v16h pa = load_a(&Pt[wave][0][0], 32);   // same-wave DS, in-order

    // ---- O += P * V  (V pre-transposed: contiguous along keys) ----
#pragma unroll
    for (int d = 0; d < 4; ++d) {
      const f16* Vb = Vt + (((size_t)(b * NH + h) * DK + d * 16)) * LE + kc;
      O[d] = wmma_f16(pa, load_b(Vb, LE), O[d]);
    }
  }

  // ---- normalize and store to concat layout [b, l, h*64+d] ----
#pragma unroll
  for (int r = 0; r < 8; ++r) {
    const float inv = 1.0f / lrow[r];
    const size_t rowoff = ((size_t)(b * LD + q0 + mb + r)) * HID + h * DK;
#pragma unroll
    for (int d = 0; d < 4; ++d)
      A16[rowoff + d * 16 + ncol] = (f16)(O[d][r] * inv);
  }
}

// ---------------- layernorm (unbiased variance, eps=1e-8) ----------------
__global__ void __launch_bounds__(256)
ln_kernel(const float* __restrict__ res, const float* __restrict__ gamma,
          const float* __restrict__ beta, float* __restrict__ out) {
  const int row = blockIdx.x;
  const float* x = res + (size_t)row * HID;
  const int tid = threadIdx.x;

  float s = 0.0f, sq = 0.0f;
  for (int i = tid; i < HID; i += 256) {
    const float v = x[i];
    s += v;
    sq += v * v;
  }
  __shared__ float sh[256], sh2[256];
  sh[tid] = s; sh2[tid] = sq;
  __syncthreads();
  for (int st = 128; st > 0; st >>= 1) {
    if (tid < st) { sh[tid] += sh[tid + st]; sh2[tid] += sh2[tid + st]; }
    __syncthreads();
  }
  const float mean = sh[0] / (float)HID;
  const float var  = (sh2[0] - (float)HID * mean * mean) / (float)(HID - 1);
  const float rstd = rsqrtf(var + EPS);
  const float g = gamma[0], be = beta[0];
  for (int i = tid; i < HID; i += 256)
    out[(size_t)row * HID + i] = g * (x[i] - mean) * rstd + be;
}

// ---------------- host-side orchestration ----------------
extern "C" void kernel_launch(void* const* d_in, const int* in_sizes, int n_in,
                              void* d_out, int out_size, void* d_ws, size_t ws_size,
                              hipStream_t stream) {
  const float* en   = (const float*)d_in[0];
  const float* de   = (const float*)d_in[1];
  const unsigned char* mask = (const unsigned char*)d_in[2];
  const float* wq = (const float*)d_in[3];
  const float* bq = (const float*)d_in[4];
  const float* wk = (const float*)d_in[5];
  const float* bk = (const float*)d_in[6];
  const float* wv = (const float*)d_in[7];
  const float* bv = (const float*)d_in[8];
  const float* wo = (const float*)d_in[9];
  const float* bo = (const float*)d_in[10];
  const float* gamma = (const float*)d_in[11];
  const float* beta  = (const float*)d_in[12];
  float* out = (float*)d_out;

  // workspace carve-up (256B aligned)
  char* base = (char*)d_ws;
  size_t off = 0;
  auto carve = [&](size_t bytes) {
    char* p = base + off;
    off = (off + bytes + 255) & ~(size_t)255;
    return p;
  };
  const size_t actHalves = (size_t)BB * LD * HID;  // de-sized (LD == LE)
  f16* de16 = (f16*)carve(actHalves * sizeof(f16));
  f16* en16 = (f16*)carve(actHalves * sizeof(f16));
  f16* wq16 = (f16*)carve((size_t)HID * HID * sizeof(f16));
  f16* wk16 = (f16*)carve((size_t)HID * HID * sizeof(f16));
  f16* wv16 = (f16*)carve((size_t)HID * HID * sizeof(f16));
  f16* wo16 = (f16*)carve((size_t)HID * HID * sizeof(f16));
  f16* Qc   = (f16*)carve(actHalves * sizeof(f16));
  f16* Kc   = (f16*)carve(actHalves * sizeof(f16));
  f16* Vt   = (f16*)carve(actHalves * sizeof(f16));
  f16* A16  = (f16*)carve(actHalves * sizeof(f16));
  float* res = (float*)carve(actHalves * sizeof(float));
  (void)ws_size; (void)in_sizes; (void)n_in; (void)out_size;

  const int nAct = BB * LD * HID;        // 6,291,456
  const int nW   = HID * HID;            // 589,824
  cvt_f32_f16<<<(nAct + 255) / 256, 256, 0, stream>>>(de, de16, nAct);
  cvt_f32_f16<<<(nAct + 255) / 256, 256, 0, stream>>>(en, en16, nAct);
  cvt_f32_f16<<<(nW + 255) / 256, 256, 0, stream>>>(wq, wq16, nW);
  cvt_f32_f16<<<(nW + 255) / 256, 256, 0, stream>>>(wk, wk16, nW);
  cvt_f32_f16<<<(nW + 255) / 256, 256, 0, stream>>>(wv, wv16, nW);
  cvt_f32_f16<<<(nW + 255) / 256, 256, 0, stream>>>(wo, wo16, nW);

  dim3 gemmGrid(BB * LD / 16, HID / 128);   // (512, 6)
  dim3 gemmBlk(256);
  // Q = de @ wq^T + bq   (head-reordered f16)
  proj_gemm<0><<<gemmGrid, gemmBlk, 0, stream>>>(de16, wq16, bq, nullptr, Qc, nullptr, LD);
  // K = en @ wk^T + bk
  proj_gemm<0><<<gemmGrid, gemmBlk, 0, stream>>>(en16, wk16, bk, nullptr, Kc, nullptr, LE);
  // V = en @ wv^T + bv   (per-head transposed f16)
  proj_gemm<1><<<gemmGrid, gemmBlk, 0, stream>>>(en16, wv16, bv, nullptr, Vt, nullptr, LE);

  // attention: B*NH*(LD/16) waves, 8 waves/block
  attn_kernel<<<(BB * NH * (LD / 16)) / 8, 256, 0, stream>>>(Qc, Kc, Vt, mask, A16);

  // res = attn @ wo^T + bo + de   (fp32)
  proj_gemm<2><<<gemmGrid, gemmBlk, 0, stream>>>(A16, wo16, bo, de, nullptr, res, LD);

  // layernorm -> output
  ln_kernel<<<BB * LD, 256, 0, stream>>>(res, gamma, beta, out);
}
